// ImprovedMoELayer_14998025797843
// MI455X (gfx1250) — matrix operand order repaired and checked by reading, compiled, not run on hardware
//
#include <hip/hip_runtime.h>
#include <hip/hip_bf16.h>
#include <math.h>

#define N_TOK 16384
#define DIM   1024
#define DFF   4096
#define NEXP  8
#define CAP   2560
#define LDS_STRIDE 40     // bf16 elements per LDS row: 32 data + 8 pad (80B, 16B aligned)
#define TILE_ELEMS (128 * LDS_STRIDE)

typedef __attribute__((ext_vector_type(16))) __bf16 v16bf;
typedef __attribute__((ext_vector_type(8)))  __bf16 v8bf;
typedef __attribute__((ext_vector_type(8)))  float  v8f;

// ------------------------------------------------------- async global->LDS
#if defined(__gfx1250__) && __has_builtin(__builtin_amdgcn_global_load_async_to_lds_b128)
#define HAVE_ASYNC 1
#else
#define HAVE_ASYNC 0
#endif

#if HAVE_ASYNC
typedef int v4i __attribute__((vector_size(16)));        // matches builtin param type
typedef __attribute__((address_space(1))) v4i* gbl_v4i_p; // prints as "int4 __device__ *"
typedef __attribute__((address_space(3))) v4i* lds_v4i_p; // LDS address space
__device__ __forceinline__ void async_cp16(const void* g, void* l) {
  // 16B per lane, memory -> LDS, tracked by ASYNCcnt (no VGPR data round-trip)
  __builtin_amdgcn_global_load_async_to_lds_b128(
      (gbl_v4i_p)(uintptr_t)g, (lds_v4i_p)(unsigned)(uintptr_t)l, 0, 0);
}
#endif

template <int N>
__device__ __forceinline__ void wait_asynccnt() {
#if __has_builtin(__builtin_amdgcn_s_wait_asynccnt)
  __builtin_amdgcn_s_wait_asynccnt((short)N);
#else
  asm volatile("s_wait_asynccnt %0" :: "i"(N) : "memory");
#endif
}

// ---------------------------------------------------------------- fragment load
// Build a 16x32 (A) or 32x16 (B) bf16 WMMA fragment from two contiguous 16B
// LDS chunks, per the CDNA5 16-bit operand VGPR layout.
__device__ __forceinline__ v16bf load_frag(const __bf16* p, int off2) {
  v8bf lo = *(const v8bf*)p;
  v8bf hi = *(const v8bf*)(p + off2);
  v16bf r;
#pragma unroll
  for (int i = 0; i < 8; ++i) { r[i] = lo[i]; r[i + 8] = hi[i]; }
  return r;
}

// ---------------------------------------------------------------- init kernels
__global__ void init_misc_kernel(int* __restrict__ tok, float* __restrict__ scal) {
  int i = blockIdx.x * 256 + threadIdx.x;
  if (i < NEXP * CAP) tok[i] = -1;
  if (i < 16) scal[i] = 0.0f;
}

__global__ void zero_buf_kernel(uint4* __restrict__ p, size_t n4) {
  size_t i = (size_t)blockIdx.x * 256 + threadIdx.x;
  if (i < n4) p[i] = make_uint4(0u, 0u, 0u, 0u);
}

// ------------------------------------------- weight transpose-convert (once)
// w (K x N f32 per expert, row-major) -> wt (N x K bf16 per expert) through a
// padded LDS tile so both global read and global write stay coalesced.
__global__ __launch_bounds__(256)
void convert_wt_kernel(const float* __restrict__ w, __bf16* __restrict__ wt,
                       int K, int N) {
  __shared__ __bf16 tile[64][72];            // stride 144B (16B aligned rows)
  const int e = blockIdx.z;
  const int k0 = blockIdx.x * 64, n0 = blockIdx.y * 64;
  w  += (size_t)e * K * N;
  wt += (size_t)e * N * K;
  for (int s = threadIdx.x; s < 64 * 16; s += 256) {
    const int r = s >> 4, c4 = s & 15;       // k-row, float4 column
    float4 v = *(const float4*)(w + (size_t)(k0 + r) * N + n0 + c4 * 4);
    tile[c4 * 4 + 0][r] = (__bf16)v.x;
    tile[c4 * 4 + 1][r] = (__bf16)v.y;
    tile[c4 * 4 + 2][r] = (__bf16)v.z;
    tile[c4 * 4 + 3][r] = (__bf16)v.w;
  }
  __syncthreads();
  for (int s = threadIdx.x; s < 64 * 8; s += 256) {
    const int n = s >> 3, c8 = s & 7;        // n-row, 8-bf16 chunk along k
    v8bf v = *(const v8bf*)&tile[n][c8 * 8];
    *(v8bf*)(wt + (size_t)(n0 + n) * K + k0 + c8 * 8) = v;
  }
}

// ---------------------------------------------------------------- gating
// One wave32 per token: logits = x . gate_w^T (E=8), softmax/argmax/logsumexp.
__global__ __launch_bounds__(256)
void gate_kernel(const float* __restrict__ x, const float* __restrict__ gw,
                 int* __restrict__ eidx, float* __restrict__ scal) {
  __shared__ float gws[NEXP * DIM];          // 32 KB of the 320 KB WGP LDS
  for (int i = threadIdx.x; i < NEXP * DIM; i += 256) gws[i] = gw[i];
  __syncthreads();

  const int wave = threadIdx.x >> 5, lane = threadIdx.x & 31;
  const int t = blockIdx.x * 8 + wave;
  const float* xr = x + (size_t)t * DIM;

  float acc[NEXP];
#pragma unroll
  for (int e = 0; e < NEXP; ++e) acc[e] = 0.0f;

  for (int d = lane * 4; d < DIM; d += 128) {
    float4 xv = *(const float4*)(xr + d);
#pragma unroll
    for (int e = 0; e < NEXP; ++e) {
      const float* g = &gws[e * DIM + d];
      acc[e] += xv.x * g[0] + xv.y * g[1] + xv.z * g[2] + xv.w * g[3];
    }
  }
#pragma unroll
  for (int off = 16; off > 0; off >>= 1)
#pragma unroll
    for (int e = 0; e < NEXP; ++e) acc[e] += __shfl_xor(acc[e], off, 32);

  __shared__ float wlz[8];
  __shared__ float wp[8][NEXP];
  if (lane == 0) {
    float m = acc[0]; int mi = 0;
#pragma unroll
    for (int e = 1; e < NEXP; ++e) if (acc[e] > m) { m = acc[e]; mi = e; }
    float s = 0.0f, ex[NEXP];
#pragma unroll
    for (int e = 0; e < NEXP; ++e) { ex[e] = __expf(acc[e] - m); s += ex[e]; }
    const float logz = m + __logf(s);
    eidx[t] = mi;
    wlz[wave] = logz * logz;
    const float inv = 1.0f / s;
#pragma unroll
    for (int e = 0; e < NEXP; ++e) wp[wave][e] = ex[e] * inv;
  }
  __syncthreads();
  if (threadIdx.x == 0) {
    float s = 0.0f;
    for (int w = 0; w < 8; ++w) s += wlz[w];
    atomicAdd(&scal[0], s);
  }
  if (threadIdx.x < NEXP) {
    float s = 0.0f;
    for (int w = 0; w < 8; ++w) s += wp[w][threadIdx.x];
    atomicAdd(&scal[1 + threadIdx.x], s);
  }
}

// ---------------------------------------------------------------- stable ranking
__global__ __launch_bounds__(256)
void rank_kernel(const int* __restrict__ eidx, int* __restrict__ rank,
                 int* __restrict__ tok, int* __restrict__ counts) {
  __shared__ int base[NEXP];
  __shared__ int cnt[8][NEXP];
  const int tid = threadIdx.x, wave = tid >> 5, lane = tid & 31;
  if (tid < NEXP) base[tid] = 0;
  __syncthreads();

  for (int c = 0; c < N_TOK; c += 256) {
    const int t = c + tid;
    const int e = eidx[t];
    unsigned same = 0u; int myCnt = 0;
#pragma unroll
    for (int ee = 0; ee < NEXP; ++ee) {
      unsigned b = (unsigned)__ballot(e == ee);   // wave32: low 32 bits
      if (e == ee) same = b;
      if (lane == ee) myCnt = __popc(b);
    }
    if (lane < NEXP) cnt[wave][lane] = myCnt;
    __syncthreads();
    int pre = 0;
    for (int w = 0; w < wave; ++w) pre += cnt[w][e];
    const int r = base[e] + pre + __popc(same & ((1u << lane) - 1u));
    rank[t] = r;
    if (r < CAP) tok[e * CAP + r] = t;
    __syncthreads();
    if (tid < NEXP) {
      int s = 0;
      for (int w = 0; w < 8; ++w) s += cnt[w][tid];
      base[tid] += s;
    }
    __syncthreads();
  }
  if (tid < NEXP) counts[tid] = base[tid];
}

// ---------------------------------------------------------------- dispatch
__global__ __launch_bounds__(256)
void dispatch_kernel(const float* __restrict__ x, const int* __restrict__ eidx,
                     const int* __restrict__ rank, float* __restrict__ out,
                     __bf16* __restrict__ buf) {
  const size_t idx = (size_t)blockIdx.x * 256 + threadIdx.x;  // one float4
  const int t  = (int)(idx >> 8);       // DIM/4 = 256 float4 per token
  const int c4 = (int)(idx & 255);
  float4 v = *(const float4*)(x + idx * 4);
  *(float4*)(out + idx * 4) = v;
  const int r = rank[t];
  if (r < CAP) {
    const int e = eidx[t];
    __bf16* d = buf + ((size_t)e * CAP + r) * DIM + c4 * 4;
    d[0] = (__bf16)v.x; d[1] = (__bf16)v.y;
    d[2] = (__bf16)v.z; d[3] = (__bf16)v.w;
  }
}

// ---------------------------------------------------------------- WMMA GEMM
// A row-major (M x K), B pre-transposed N-major (N x K): both tiles are plain
// contiguous 32B row copies -> async global->LDS with double buffering.
// MODE 1: H = gelu(A @ B^T' + bias) -> bf16 hbuf
// MODE 2: Y = A @ B^T' + bias, scattered to out rows via tok_of_slot
template <int MODE>
__global__ __launch_bounds__(256)
void moe_gemm_kernel(const __bf16* __restrict__ A, const __bf16* __restrict__ Bt,
                     const float* __restrict__ bias, __bf16* __restrict__ outH,
                     float* __restrict__ outY, const int* __restrict__ tok,
                     int Mdim, int Ndim, int Kdim) {
  __shared__ __bf16 As[2][TILE_ELEMS];   // [m][k], stride 40
  __shared__ __bf16 Bs[2][TILE_ELEMS];   // [n][k], stride 40
  const int tid = threadIdx.x;
  const int e = blockIdx.z;
  const int m0 = blockIdx.y * 128, n0 = blockIdx.x * 128;

  A    += (size_t)e * Mdim * Kdim;
  Bt   += (size_t)e * Ndim * Kdim;
  bias += (size_t)e * Ndim;
  if (MODE == 1) outH += (size_t)e * Mdim * Ndim;
  else           tok  += e * CAP;

  // staging: thread -> (row rr, k-half rh); 2x16B for A, 2x16B for B
  const int rr = tid >> 1, rh = tid & 1;
  const __bf16* agp = A  + (size_t)(m0 + rr) * Kdim + rh * 16;
  const __bf16* bgp = Bt + (size_t)(n0 + rr) * Kdim + rh * 16;
  const int lofs = rr * LDS_STRIDE + rh * 16;

  const int wave = tid >> 5, lane = tid & 31;
  const int half = lane >> 4, l15 = lane & 15;
  const int wm = (wave & 1) * 64, wn = (wave >> 1) * 32;

  v8f acc[4][2] = {};

  // ---- stage tile 0
#if HAVE_ASYNC
  async_cp16(agp,     &As[0][lofs]);
  async_cp16(agp + 8, &As[0][lofs + 8]);
  async_cp16(bgp,     &Bs[0][lofs]);
  async_cp16(bgp + 8, &Bs[0][lofs + 8]);
#else
  {
    uint4 a0 = *(const uint4*)agp, a1 = *(const uint4*)(agp + 8);
    uint4 b0 = *(const uint4*)bgp, b1 = *(const uint4*)(bgp + 8);
    *(uint4*)&As[0][lofs] = a0;  *(uint4*)&As[0][lofs + 8] = a1;
    *(uint4*)&Bs[0][lofs] = b0;  *(uint4*)&Bs[0][lofs + 8] = b1;
  }
#endif
  agp += 32; bgp += 32;

  int cur = 0;
  for (int k0 = 0; k0 < Kdim; k0 += 32) {
    const bool more = (k0 + 32) < Kdim;
    if (more) {   // prefetch next K-tile into the other buffer
#if HAVE_ASYNC
      async_cp16(agp,     &As[cur ^ 1][lofs]);
      async_cp16(agp + 8, &As[cur ^ 1][lofs + 8]);
      async_cp16(bgp,     &Bs[cur ^ 1][lofs]);
      async_cp16(bgp + 8, &Bs[cur ^ 1][lofs + 8]);
#else
      uint4 a0 = *(const uint4*)agp, a1 = *(const uint4*)(agp + 8);
      uint4 b0 = *(const uint4*)bgp, b1 = *(const uint4*)(bgp + 8);
      *(uint4*)&As[cur ^ 1][lofs] = a0;  *(uint4*)&As[cur ^ 1][lofs + 8] = a1;
      *(uint4*)&Bs[cur ^ 1][lofs] = b0;  *(uint4*)&Bs[cur ^ 1][lofs + 8] = b1;
#endif
      agp += 32; bgp += 32;
    }
#if HAVE_ASYNC
    // async loads complete in order: <=4 outstanding means current tile landed
    if (more) wait_asynccnt<4>(); else wait_asynccnt<0>();
#endif
    __syncthreads();

    v16bf af[4], bfr[2];
#pragma unroll
    for (int mf = 0; mf < 4; ++mf)   // A frag: lane<16 K{0..7,16..23}, lane>=16 K{8..15,24..31}
      af[mf] = load_frag(&As[cur][(wm + mf * 16 + l15) * LDS_STRIDE + half * 8], 16);
#pragma unroll
    for (int nf = 0; nf < 2; ++nf)   // B frag: lane<16 K{0..7,16..23}, lane>=16 K{8..15,24..31}
      bfr[nf] = load_frag(&Bs[cur][(wn + nf * 16 + l15) * LDS_STRIDE + half * 8], 16);
#pragma unroll
    for (int mf = 0; mf < 4; ++mf)
#pragma unroll
      for (int nf = 0; nf < 2; ++nf)
        acc[mf][nf] = __builtin_amdgcn_wmma_f32_16x16x32_bf16(
            false, af[mf], false, bfr[nf], (short)0, acc[mf][nf], false, false);

    __syncthreads();   // all reads of buf `cur` done before it becomes a write target
    cur ^= 1;
  }

  // epilogue: C layout — VGPR r: lanes 0-15 -> M=r, lanes 16-31 -> M=r+8; N=lane&15
#pragma unroll
  for (int mf = 0; mf < 4; ++mf) {
#pragma unroll
    for (int nf = 0; nf < 2; ++nf) {
      const int col = n0 + wn + nf * 16 + l15;
      const float bv = bias[col];
#pragma unroll
      for (int r = 0; r < 8; ++r) {
        const int row = m0 + wm + mf * 16 + r + 8 * half;
        float v = acc[mf][nf][r] + bv;
        if (MODE == 1) {
          v = 0.5f * v * (1.0f + erff(v * 0.70710678118654752f));  // exact gelu
          outH[(size_t)row * Ndim + col] = (__bf16)v;
        } else {
          const int t = tok[row];
          if (t >= 0) outY[(size_t)t * DIM + col] = v;
        }
      }
    }
  }
}

// ---------------------------------------------------------------- losses
__global__ void finalize_kernel(const float* __restrict__ scal,
                                const int* __restrict__ counts,
                                float* __restrict__ o) {
  if (threadIdx.x == 0) {
    const float invN = 1.0f / (float)N_TOK;
    float z_loss = 0.001f * scal[0] * invN;
    float aux = 0.0f;
    int disp = 0;
    for (int e = 0; e < NEXP; ++e) {
      aux += ((float)counts[e] * invN) * (scal[1 + e] * invN);
      disp += (counts[e] < CAP) ? counts[e] : CAP;
    }
    aux *= (float)NEXP;
    o[0] = z_loss;
    o[1] = aux;
    o[2] = 1.0f - (float)disp * invN;
  }
}

// ---------------------------------------------------------------- launch
extern "C" void kernel_launch(void* const* d_in, const int* in_sizes, int n_in,
                              void* d_out, int out_size, void* d_ws, size_t ws_size,
                              hipStream_t stream) {
  (void)in_sizes; (void)n_in; (void)out_size; (void)ws_size;
  const float* hidden = (const float*)d_in[0];
  const float* gate_w = (const float*)d_in[1];
  const float* w1     = (const float*)d_in[2];
  const float* b1     = (const float*)d_in[3];
  const float* w2     = (const float*)d_in[4];
  const float* b2     = (const float*)d_in[5];
  float* out = (float*)d_out;

  // workspace layout
  char* ws = (char*)d_ws;
  constexpr size_t SZ_W1T  = (size_t)NEXP * DIM * DFF * 2;   // 64 MB bf16 (N-major)
  constexpr size_t SZ_W2T  = (size_t)NEXP * DFF * DIM * 2;   // 64 MB bf16 (N-major)
  constexpr size_t SZ_BUF  = (size_t)NEXP * CAP * DIM * 2;   // 40 MB bf16
  constexpr size_t SZ_HBUF = (size_t)NEXP * CAP * DFF * 2;   // 160 MB bf16
  constexpr size_t OFF_W1T  = 0;
  constexpr size_t OFF_W2T  = OFF_W1T + SZ_W1T;
  constexpr size_t OFF_BUF  = OFF_W2T + SZ_W2T;
  constexpr size_t OFF_HBUF = OFF_BUF + SZ_BUF;
  constexpr size_t OFF_EIDX = OFF_HBUF + SZ_HBUF;
  constexpr size_t OFF_RANK = OFF_EIDX + (size_t)N_TOK * 4;
  constexpr size_t OFF_TOK  = OFF_RANK + (size_t)N_TOK * 4;
  constexpr size_t OFF_CNT  = OFF_TOK + (size_t)NEXP * CAP * 4;
  constexpr size_t OFF_SCAL = OFF_CNT + 64;

  __bf16* w1t  = (__bf16*)(ws + OFF_W1T);
  __bf16* w2t  = (__bf16*)(ws + OFF_W2T);
  __bf16* buf  = (__bf16*)(ws + OFF_BUF);
  __bf16* hbuf = (__bf16*)(ws + OFF_HBUF);
  int*   eidx   = (int*)(ws + OFF_EIDX);
  int*   rank   = (int*)(ws + OFF_RANK);
  int*   tok    = (int*)(ws + OFF_TOK);
  int*   counts = (int*)(ws + OFF_CNT);
  float* scal   = (float*)(ws + OFF_SCAL);

  // 1) init scratch (deterministic each call)
  init_misc_kernel<<<(NEXP * CAP + 255) / 256, 256, 0, stream>>>(tok, scal);
  const size_t buf4 = SZ_BUF / 16;
  zero_buf_kernel<<<(unsigned)((buf4 + 255) / 256), 256, 0, stream>>>((uint4*)buf, buf4);

  // 2) weights -> bf16, transposed to N-major for contiguous B staging
  convert_wt_kernel<<<dim3(DIM / 64, DFF / 64, NEXP), 256, 0, stream>>>(w1, w1t, DIM, DFF);
  convert_wt_kernel<<<dim3(DFF / 64, DIM / 64, NEXP), 256, 0, stream>>>(w2, w2t, DFF, DIM);

  // 3) gating + 4) ranking + 5) dispatch
  gate_kernel<<<N_TOK / 8, 256, 0, stream>>>(hidden, gate_w, eidx, scal);
  rank_kernel<<<1, 256, 0, stream>>>(eidx, rank, tok, counts);
  dispatch_kernel<<<(N_TOK * (DIM / 4)) / 256, 256, 0, stream>>>(hidden, eidx, rank, out, buf);

  // 6) expert GEMMs on the WMMA path (async double-buffered LDS pipeline)
  moe_gemm_kernel<1><<<dim3(DFF / 128, CAP / 128, NEXP), 256, 0, stream>>>(
      buf, w1t, b1, hbuf, nullptr, nullptr, CAP, DFF, DIM);
  moe_gemm_kernel<2><<<dim3(DIM / 128, CAP / 128, NEXP), 256, 0, stream>>>(
      hbuf, w2t, b2, nullptr, out, tok, CAP, DIM, DFF);

  // 7) scalar losses
  finalize_kernel<<<1, 32, 0, stream>>>(scal, counts, out + (size_t)N_TOK * DIM);
}
